// EGNNLayer_65017214927603
// MI455X (gfx1250) — matrix operand order repaired and checked by reading, not compiled, not run on hardware
//
#include <hip/hip_runtime.h>
#include <hip/hip_bf16.h>

#define HID   256
#define NN    384
#define RT    16              // receivers per block
#define SB    32              // senders per block
#define NRT   (NN / RT)       // 24
#define NSB   (NN / SB)       // 12
#define CHUNKS (SB / 2)       // 16 chunks of 2 senders -> 32 edges (2 M-tiles)
#define HSTR  (HID + 16)      // padded LDS row stride (halves)

typedef __attribute__((ext_vector_type(16))) __bf16 v16bf;
typedef __attribute__((ext_vector_type(8)))  float  v8f;

union FragB { v16bf v; uint4 q[2]; };

__device__ __forceinline__ float silu_f(float x) {
  // x * sigmoid(x) with hardware v_rcp_f32 (avoids precise-div expansion)
  return x * __builtin_amdgcn_rcpf(1.0f + __expf(-x));
}
__device__ __forceinline__ unsigned short f2bf(float f) {
  unsigned u = __builtin_bit_cast(unsigned, f);
  u += 0x7FFFu + ((u >> 16) & 1u);          // round-to-nearest-even
  return (unsigned short)(u >> 16);
}
__device__ __forceinline__ unsigned f2bf_pk(float lo, float hi) {
#if defined(__gfx1250__) && __has_builtin(__builtin_amdgcn_cvt_pk_bf16_f32)
  typedef __bf16 v2bf __attribute__((ext_vector_type(2)));
  v2bf r = __builtin_amdgcn_cvt_pk_bf16_f32(lo, hi);
  return __builtin_bit_cast(unsigned, r);
#else
  return (unsigned)f2bf(lo) | ((unsigned)f2bf(hi) << 16);
#endif
}

// ---------------------------------------------------------------------------
// Prep 1: swizzle eW2 / pW1 (f32 [N][K] row-major) into bf16 B-fragment order:
// [kt][nt][lane][p], lane's 16 halves = K = kt*32 + (lane<16?0:8) + (p&7) + (p<8?0:16),
// N = nt*16 + lane%16  ->  one contiguous 32B read per lane per fragment.
// ---------------------------------------------------------------------------
__global__ void prep_w_kernel(const float* __restrict__ eW2,
                              const float* __restrict__ pW1,
                              unsigned short* __restrict__ gW2,
                              unsigned short* __restrict__ gPW) {
  int d = blockIdx.x * blockDim.x + threadIdx.x;
  if (d >= HID * HID) return;
  int kt   = d >> 13;
  int nt   = (d >> 9) & 15;
  int idx  = d & 511;
  int lane = idx >> 4;
  int p    = idx & 15;
  int K = (kt << 5) + ((lane >= 16) ? 8 : 0) + (p & 7) + ((p >= 8) ? 16 : 0);
  int N = (nt << 4) + (lane & 15);
  gW2[d] = f2bf(eW2[N * HID + K]);
  gPW[d] = f2bf(pW1[N * HID + K]);
}

// ---------------------------------------------------------------------------
// Prep 2: A = nodes @ Wr^T + eb1, B = nodes @ Ws^T (exact f32), Wrad column.
// Exploits rank structure of layer 1: deletes the [E,513]x[513,256] GEMM.
// ---------------------------------------------------------------------------
__global__ void prep_ab_kernel(const float* __restrict__ nodes,
                               const float* __restrict__ eW1,
                               const float* __restrict__ eb1,
                               float* __restrict__ Abuf,
                               float* __restrict__ Bbuf,
                               float* __restrict__ Wrad) {
  int t = blockIdx.x * blockDim.x + threadIdx.x;
  if (t < NN * HID) {
    int r = t >> 8, h = t & 255;
    const float* w = eW1 + h * (2 * HID + 1);
    const float* x = nodes + r * HID;
    float acc = eb1[h];
    #pragma unroll 8
    for (int k = 0; k < HID; ++k) acc += x[k] * w[k];
    Abuf[t] = acc;
  } else if (t < 2 * NN * HID) {
    int u = t - NN * HID;
    int s = u >> 8, h = u & 255;
    const float* w = eW1 + h * (2 * HID + 1) + HID;
    const float* x = nodes + s * HID;
    float acc = 0.f;
    #pragma unroll 8
    for (int k = 0; k < HID; ++k) acc += x[k] * w[k];
    Bbuf[u] = acc;
  } else {
    int u = t - 2 * NN * HID;
    if (u < HID) Wrad[u] = eW1[u * (2 * HID + 1) + 2 * HID];
  }
}

// ---------------------------------------------------------------------------
// Edge kernel: block = 16 receivers x 32 senders, 8 waves.
// Weights live in registers (2x 128 VGPRs, read once from L2 per block).
// A-fragments double-buffered from LDS against the WMMA pipe.
// agg accumulates in registers; deterministic partial slabs to workspace.
// ---------------------------------------------------------------------------
__global__ __launch_bounds__(256) void edge_kernel(
    const float* __restrict__ pos,
    const float* __restrict__ Abuf, const float* __restrict__ Bbuf,
    const float* __restrict__ Wrad,
    const unsigned short* __restrict__ gW2, const unsigned short* __restrict__ gPW,
    const float* __restrict__ eb2, const float* __restrict__ pb1,
    const float* __restrict__ pW2, const float* __restrict__ pb2,
    float* __restrict__ gAggP,   // [NSB][NN][HID]
    float* __restrict__ gPosP)   // [NRT][NN][3]
{
  __shared__ float sRad[SB * RT];
  __shared__ float sPosR[RT][3];
  __shared__ float sPosS[SB][3];
  __shared__ unsigned short sH[32 * HSTR];   // h1 chunk (A-operand layout)
  __shared__ unsigned short sM[32 * HSTR];   // msg chunk
  __shared__ float sPart[8][32];             // per-wave pW2 partial dots
  __shared__ float sTrans[SB][3];

  const int tid  = threadIdx.x;
  const int wave = tid >> 5, lane = tid & 31;
  const int rt = blockIdx.x % NRT, sb = blockIdx.x / NRT;
  const int r0 = rt * RT, s0 = sb * SB;

  const int nt0 = wave * 2, nt1 = nt0 + 1;
  const int nlo = lane & 15;

  // ---- register-resident weight fragments: one L2 read per block ----
  FragB wE[2][8], wP[2][8];
  #pragma unroll
  for (int kt = 0; kt < 8; ++kt) {
    const uint4* w0 = (const uint4*)(gW2 + (kt * 16 + nt0) * 512 + lane * 16);
    const uint4* w1 = (const uint4*)(gW2 + (kt * 16 + nt1) * 512 + lane * 16);
    wE[0][kt].q[0] = w0[0]; wE[0][kt].q[1] = w0[1];
    wE[1][kt].q[0] = w1[0]; wE[1][kt].q[1] = w1[1];
    const uint4* p0 = (const uint4*)(gPW + (kt * 16 + nt0) * 512 + lane * 16);
    const uint4* p1 = (const uint4*)(gPW + (kt * 16 + nt1) * 512 + lane * 16);
    wP[0][kt].q[0] = p0[0]; wP[0][kt].q[1] = p0[1];
    wP[1][kt].q[0] = p1[0]; wP[1][kt].q[1] = p1[1];
  }

  if (tid < RT * 3) sPosR[tid / 3][tid % 3] = pos[(r0 + tid / 3) * 3 + tid % 3];
  if (tid < SB * 3) { sPosS[tid / 3][tid % 3] = pos[(s0 + tid / 3) * 3 + tid % 3];
                      sTrans[tid / 3][tid % 3] = 0.f; }
  __syncthreads();
  for (int e = tid; e < SB * RT; e += 256) {
    int ls = e >> 4, lr = e & 15;
    float dx = sPosS[ls][0] - sPosR[lr][0];
    float dy = sPosS[ls][1] - sPosR[lr][1];
    float dz = sPosS[ls][2] - sPosR[lr][2];
    sRad[e] = dx * dx + dy * dy + dz * dz;
  }
  __syncthreads();

  const float eb2n0 = eb2[nt0 * 16 + nlo], eb2n1 = eb2[nt1 * 16 + nlo];
  const float pb1n0 = pb1[nt0 * 16 + nlo], pb1n1 = pb1[nt1 * 16 + nlo];
  const float pw2n0 = pW2[nt0 * 16 + nlo], pw2n1 = pW2[nt1 * 16 + nlo];
  const int n0 = nt0 * 16 + nlo, n1 = n0 + 16;

  float aggA[8], aggB[8];                  // register agg accumulators
  #pragma unroll
  for (int v = 0; v < 8; ++v) { aggA[v] = 0.f; aggB[v] = 0.f; }

  for (int c = 0; c < CHUNKS; ++c) {
    // ---- h1 stage: 32 edges x 256 dims, f32 exact -> silu -> bf16 LDS ----
    {
      int m  = tid & 31;
      int hb = (tid >> 5) << 5;
      int lr = m & 15, ls = m >> 4;
      int r = r0 + lr, s = s0 + c * 2 + ls;
      float rad = sRad[(c * 2 + ls) * 16 + lr];
      const float4* Ap = (const float4*)(Abuf + r * HID + hb);
      const float4* Bp = (const float4*)(Bbuf + s * HID + hb);
      const float4* Wp = (const float4*)(Wrad + hb);
      unsigned short* Hp = sH + m * HSTR + hb;
      #pragma unroll
      for (int q = 0; q < 8; ++q) {
        float4 a = Ap[q], b = Bp[q], w = Wp[q];
        unsigned lo = f2bf_pk(silu_f(a.x + b.x + rad * w.x),
                              silu_f(a.y + b.y + rad * w.y));
        unsigned hi = f2bf_pk(silu_f(a.z + b.z + rad * w.z),
                              silu_f(a.w + b.w + rad * w.w));
        *(uint2*)(Hp + q * 4) = make_uint2(lo, hi);
      }
    }
    __syncthreads();

    // ---- layer-2 GEMM: msg = silu(h1 @ eW2^T + eb2) ----
    #pragma unroll
    for (int mt = 0; mt < 2; ++mt) {
      const uint4* hp = (const uint4*)(sH + (mt * 16 + nlo) * HSTR)
                        + ((lane < 16) ? 0 : 1);
      v8f acc0, acc1;
      #pragma unroll
      for (int z = 0; z < 8; ++z) { acc0[z] = 0.f; acc1[z] = 0.f; }
      FragB acur, anext;
      acur.q[0] = hp[0]; acur.q[1] = hp[2];
      #pragma unroll
      for (int kt = 0; kt < 8; ++kt) {
        if (kt < 7) { anext.q[0] = hp[4 * (kt + 1)];
                      anext.q[1] = hp[4 * (kt + 1) + 2]; }
        acc0 = __builtin_amdgcn_wmma_f32_16x16x32_bf16(false, acur.v, false, wE[0][kt].v,
                                                       (short)0, acc0, false, false);
        acc1 = __builtin_amdgcn_wmma_f32_16x16x32_bf16(false, acur.v, false, wE[1][kt].v,
                                                       (short)0, acc1, false, false);
        if (kt < 7) acur = anext;
      }
      #pragma unroll
      for (int v = 0; v < 8; ++v) {
        int lr = (lane < 16) ? v : v + 8;
        int m  = mt * 16 + lr;
        bool diag = (s0 + c * 2 + mt) == (r0 + lr);
        float y0 = silu_f(acc0[v] + eb2n0);
        float y1 = silu_f(acc1[v] + eb2n1);
        sM[m * HSTR + n0] = f2bf(y0);
        sM[m * HSTR + n1] = f2bf(y1);
        if (!diag) { aggA[v] += y0; aggB[v] += y1; }
      }
    }
    __syncthreads();

    // ---- pos-layer GEMM: p1 = silu(msg @ pW1^T + pb1); dot with pW2 ----
    #pragma unroll
    for (int mt = 0; mt < 2; ++mt) {
      const uint4* hp = (const uint4*)(sM + (mt * 16 + nlo) * HSTR)
                        + ((lane < 16) ? 0 : 1);
      v8f acc0, acc1;
      #pragma unroll
      for (int z = 0; z < 8; ++z) { acc0[z] = 0.f; acc1[z] = 0.f; }
      FragB acur, anext;
      acur.q[0] = hp[0]; acur.q[1] = hp[2];
      #pragma unroll
      for (int kt = 0; kt < 8; ++kt) {
        if (kt < 7) { anext.q[0] = hp[4 * (kt + 1)];
                      anext.q[1] = hp[4 * (kt + 1) + 2]; }
        acc0 = __builtin_amdgcn_wmma_f32_16x16x32_bf16(false, acur.v, false, wP[0][kt].v,
                                                       (short)0, acc0, false, false);
        acc1 = __builtin_amdgcn_wmma_f32_16x16x32_bf16(false, acur.v, false, wP[1][kt].v,
                                                       (short)0, acc1, false, false);
        if (kt < 7) acur = anext;
      }
      #pragma unroll
      for (int v = 0; v < 8; ++v) {
        int m = mt * 16 + ((lane < 16) ? v : v + 8);
        float t = silu_f(acc0[v] + pb1n0) * pw2n0 + silu_f(acc1[v] + pb1n1) * pw2n1;
        #pragma unroll
        for (int off = 1; off < 16; off <<= 1) t += __shfl_xor(t, off, 32);
        if (nlo == 0) sPart[wave][m] = t;
      }
    }
    __syncthreads();

    // ---- pos deltas: scale[m] = pb2 + sum_w sPart; clip; sum receivers ----
    if (tid < 6) {
      int ls = tid / 3, d = tid % 3;
      int sl = c * 2 + ls;
      float ps = sPosS[sl][d];
      float acc = 0.f;
      for (int lr = 0; lr < 16; ++lr) {
        int m = ls * 16 + lr;
        float sc = pb2[0];
        #pragma unroll
        for (int w = 0; w < 8; ++w) sc += sPart[w][m];
        float tr = (ps - sPosR[lr][d]) * sc;       // diagonal -> 0 automatically
        tr = fminf(fmaxf(tr, -100.f), 100.f);
        acc += tr;
      }
      sTrans[sl][d] = acc;
    }
    __syncthreads();
  }

  // ---- flush deterministic partials (each element owned by one thread) ----
  #pragma unroll
  for (int v = 0; v < 8; ++v) {
    int lr = (lane < 16) ? v : v + 8;
    gAggP[(sb * NN + r0 + lr) * HID + n0] = aggA[v];
    gAggP[(sb * NN + r0 + lr) * HID + n1] = aggB[v];
  }
  if (tid < SB * 3)
    gPosP[(rt * NN + s0 + tid / 3) * 3 + tid % 3] = sTrans[tid / 3][tid % 3];
}

// ---------------------------------------------------------------------------
// Node kernel: reduce partials in fixed order, node MLP + residual.
// ---------------------------------------------------------------------------
__global__ __launch_bounds__(256) void node_kernel(
    const float* __restrict__ nodes, const float* __restrict__ pos,
    const float* __restrict__ gAggP, const float* __restrict__ gPosP,
    const float* __restrict__ nW1, const float* __restrict__ nb1,
    const float* __restrict__ nW2, const float* __restrict__ nb2,
    float* __restrict__ outN, float* __restrict__ outP)
{
  __shared__ float sT[HID];
  __shared__ float sA[HID];
  int i = blockIdx.x, h = threadIdx.x;
  float ag = 0.f;
  for (int p = 0; p < NSB; ++p) ag += gAggP[(p * NN + i) * HID + h];
  sA[h] = ag;
  if (h < 3) {
    float pp = pos[i * 3 + h];
    for (int p = 0; p < NRT; ++p) pp += gPosP[(p * NN + i) * 3 + h];
    outP[i * 3 + h] = pp;
  }
  __syncthreads();
  const float* x = nodes + i * HID;
  const float* w = nW1 + h * (2 * HID);
  float acc = nb1[h];
  #pragma unroll 8
  for (int k = 0; k < HID; ++k) acc += x[k] * w[k];
  #pragma unroll 8
  for (int k = 0; k < HID; ++k) acc += sA[k] * w[HID + k];
  sT[h] = silu_f(acc);
  __syncthreads();
  const float* w2 = nW2 + h * HID;
  float acc2 = nb2[h];
  #pragma unroll 8
  for (int k = 0; k < HID; ++k) acc2 += sT[k] * w2[k];
  outN[i * HID + h] = x[h] + acc2;
}

// ---------------------------------------------------------------------------
extern "C" void kernel_launch(void* const* d_in, const int* in_sizes, int n_in,
                              void* d_out, int out_size, void* d_ws, size_t ws_size,
                              hipStream_t stream) {
  (void)in_sizes; (void)n_in; (void)out_size; (void)ws_size;
  const float* nodes = (const float*)d_in[0];
  const float* pos   = (const float*)d_in[1];
  const float* eW1   = (const float*)d_in[2];
  const float* eb1   = (const float*)d_in[3];
  const float* eW2   = (const float*)d_in[4];
  const float* eb2   = (const float*)d_in[5];
  const float* nW1   = (const float*)d_in[6];
  const float* nb1   = (const float*)d_in[7];
  const float* nW2   = (const float*)d_in[8];
  const float* nb2   = (const float*)d_in[9];
  const float* pW1   = (const float*)d_in[10];
  const float* pb1   = (const float*)d_in[11];
  const float* pW2   = (const float*)d_in[12];
  const float* pb2   = (const float*)d_in[13];

  char* ws = (char*)d_ws;
  unsigned short* gW2  = (unsigned short*)(ws);                  // 128 KB
  unsigned short* gPW  = (unsigned short*)(ws + 131072);         // 128 KB
  float* Abuf  = (float*)(ws + 262144);                          // 384 KB
  float* Bbuf  = (float*)(ws + 655360);                          // 384 KB
  float* Wrad  = (float*)(ws + 1048576);                         // 1 KB
  float* gAggP = (float*)(ws + 1049600);                         // 12*384*256*4
  float* gPosP = (float*)(ws + 1049600 + (size_t)NSB * NN * HID * 4);

  float* outN = (float*)d_out;                 // 384*256
  float* outP = (float*)d_out + NN * HID;      // 384*3

  prep_w_kernel<<<(HID * HID + 255) / 256, 256, 0, stream>>>(eW2, pW1, gW2, gPW);
  prep_ab_kernel<<<(2 * NN * HID + HID + 255) / 256, 256, 0, stream>>>(
      nodes, eW1, eb1, Abuf, Bbuf, Wrad);
  edge_kernel<<<NRT * NSB, 256, 0, stream>>>(
      pos, Abuf, Bbuf, Wrad, gW2, gPW, eb2, pb1, pW2, pb2, gAggP, gPosP);
  node_kernel<<<NN, 256, 0, stream>>>(
      nodes, pos, gAggP, gPosP, nW1, nb1, nW2, nb2, outN, outP);
}
